// TriXCUDAEngine_36369783063303
// MI455X (gfx1250) — compile-verified, matrix operand order checked by reading, not run on hardware
//
#include <hip/hip_runtime.h>
#include <stdint.h>

// ---------------------------------------------------------------------------
// Batched 32-bit ripple adder == out[n] = f32((a_int + b_int) mod 2^32).
// Read 64 bit-planes (f32 {0,1}), pack to u8 via v_perm_b32 byte gathers,
// reduce exactly with V_WMMA_I32_16X16X64_IU8 (byte-group partial sums),
// recombine in u32 (wrap == mod 2^32), convert once to f32.
// Memory-bound: ~260 B/element, ~545 MB total -> ~23 us at 23.3 TB/s.
// Two tiles per iteration: 64 saddr-form loads in flight before any
// dependent ALU; v_perm packing is 4 VALU per 4 bytes.
// ---------------------------------------------------------------------------

typedef __attribute__((ext_vector_type(8))) int v8i;

#define RB_BITS 32

// K-base per operand VGPR v for a lane-half (ISA 7.12.2, 8-bit 16x64 layout).
__device__ __forceinline__ int kbase_of(int v, int half) {
    return 16 * (v >> 1) + 8 * half + 4 * (v & 1);
}

// Issue all 32 loads for one 16-element tile (no dependent ALU in between).
__device__ __forceinline__ void load_tile(const uint32_t* __restrict__ a,
                                          const uint32_t* __restrict__ b,
                                          int N, int half, int n,
                                          uint32_t raw[32])
{
#pragma unroll
    for (int v = 0; v < 8; ++v) {
        const int kb = kbase_of(v, half);
#pragma unroll
        for (int j = 0; j < 4; ++j) {
            const int k = kb + j;
            const uint32_t* __restrict__ pb =
                ((k < 32) ? a : b) + (size_t)(k & 31) * (size_t)N;
            raw[4 * v + j] = pb[n];          // saddr-form global_load_b32
        }
    }
}

// Pack 32 f32 {0,1} words into 8 B-operand words: gather byte2 of each word
// (0x80 for 1.0f, 0x00 for 0.0f) with v_perm_b32, then >>7 -> 0x01/0x00
// bytes (low 7 bits of each gathered byte are zero, so no mask needed).
__device__ __forceinline__ v8i pack_tile(const uint32_t raw[32])
{
    v8i B;
#pragma unroll
    for (int v = 0; v < 8; ++v) {
        const uint32_t r0 = raw[4 * v + 0], r1 = raw[4 * v + 1];
        const uint32_t r2 = raw[4 * v + 2], r3 = raw[4 * v + 3];
        // p01 bytes: [r0.b2, r1.b2, x, x]   (sel<4 -> lo=src1, sel>=4 -> hi=src0)
        const uint32_t p01 = __builtin_amdgcn_perm(r1, r0, 0x06020602u);
        const uint32_t p23 = __builtin_amdgcn_perm(r3, r2, 0x06020602u);
        // w bytes: [p01.b0, p01.b1, p23.b0, p23.b1]
        const uint32_t w = __builtin_amdgcn_perm(p23, p01, 0x05040100u);
        B[v] = (int)(w >> 7);                // 0x80 bytes -> 0x01 bytes
    }
    return B;
}

__global__ __launch_bounds__(256)
void ripple_add_wmma_kernel(const uint32_t* __restrict__ a,
                            const uint32_t* __restrict__ b,
                            float* __restrict__ out,
                            int N, int nTiles)
{
    const int lane = threadIdx.x & 31;            // wave32
    const int wid  = (int)((blockIdx.x * blockDim.x + threadIdx.x) >> 5);
    const int nWaves = (int)((gridDim.x * blockDim.x) >> 5);
    const int half = lane >> 4;
    const int col  = lane & 15;

    // ---- Constant A (weights): A[m][k] = 2^(k&7) if (k&31)>>3 == m (m<4) ---
    v8i A;
#pragma unroll
    for (int v = 0; v < 8; ++v) {
        const int kb = kbase_of(v, half);
        uint32_t w = 0u;
#pragma unroll
        for (int j = 0; j < 4; ++j) {
            const int k = kb + j;
            const uint32_t byte = (((k & 31) >> 3) == col) ? (1u << (k & 7)) : 0u;
            w |= byte << (8 * j);
        }
        A[v] = (int)w;
    }

    // ---- Contiguous tile range per wave (sequential HBM streaming) --------
    const int tilesPerWave = (nTiles + nWaves - 1) / nWaves;
    const int t0 = wid * tilesPerWave;
    int t1 = t0 + tilesPerWave;
    if (t1 > nTiles) t1 = nTiles;

    const v8i Cz = {};
    int tile = t0;

    // ---- Main loop: two tiles per iteration (64 loads in flight) ----------
    for (; tile + 2 <= t1; tile += 2) {
        int n0 = tile * 16 + col;       if (n0 >= N) n0 = N - 1;
        int n1 = (tile + 1) * 16 + col; if (n1 >= N) n1 = N - 1;

        uint32_t raw0[32], raw1[32];
        load_tile(a, b, N, half, n0, raw0);
        load_tile(a, b, N, half, n1, raw1);

        // Prefetch ahead in this wave's sequential stream (global_prefetch_b8).
        if (n1 + 64 < N) {
            const uint32_t* pb0 = a + (size_t)(8 * half) * (size_t)N;
            __builtin_prefetch(pb0 + n1 + 64, 0, 1);
        }

        const v8i B0 = pack_tile(raw0);
        const v8i B1 = pack_tile(raw1);

        // (sgn_a, A, sgn_b, B, C, reuse_a, reuse_b)
        const v8i D0 = __builtin_amdgcn_wmma_i32_16x16x64_iu8(false, A, false, B0,
                                                              Cz, false, false);
        const v8i D1 = __builtin_amdgcn_wmma_i32_16x16x64_iu8(false, A, false, B1,
                                                              Cz, false, false);

        // Rows 0..3 (byte-group partials) live in d.s[0..3] of lanes 0..15.
        // u32 wrap on the <<24 term == dropped carry-out mod 2^32.
        if (lane < 16) {
            const int o0 = tile * 16 + lane;
            const int o1 = (tile + 1) * 16 + lane;
            const uint32_t r0 = (uint32_t)D0[0] + ((uint32_t)D0[1] << 8)
                              + ((uint32_t)D0[2] << 16) + ((uint32_t)D0[3] << 24);
            const uint32_t r1 = (uint32_t)D1[0] + ((uint32_t)D1[1] << 8)
                              + ((uint32_t)D1[2] << 16) + ((uint32_t)D1[3] << 24);
            if (o0 < N) out[o0] = (float)r0;
            if (o1 < N) out[o1] = (float)r1;
        }
    }

    // ---- Tail: at most one tile ------------------------------------------
    for (; tile < t1; ++tile) {
        int n = tile * 16 + col; if (n >= N) n = N - 1;
        uint32_t raw[32];
        load_tile(a, b, N, half, n, raw);
        const v8i B = pack_tile(raw);
        const v8i D = __builtin_amdgcn_wmma_i32_16x16x64_iu8(false, A, false, B,
                                                             Cz, false, false);
        if (lane < 16) {
            const int on = tile * 16 + lane;
            if (on < N) {
                const uint32_t res = (uint32_t)D[0] + ((uint32_t)D[1] << 8)
                                   + ((uint32_t)D[2] << 16) + ((uint32_t)D[3] << 24);
                out[on] = (float)res;
            }
        }
    }
}

extern "C" void kernel_launch(void* const* d_in, const int* in_sizes, int n_in,
                              void* d_out, int out_size, void* d_ws, size_t ws_size,
                              hipStream_t stream)
{
    const uint32_t* a = (const uint32_t*)d_in[0]; // [32, N] f32 {0,1} bit-planes
    const uint32_t* b = (const uint32_t*)d_in[1]; // [32, N]
    float* out = (float*)d_out;                   // [N] f32

    const int N = in_sizes[0] / RB_BITS;
    const int nTiles = (N + 15) / 16;

    // 512 blocks x 256 threads = 4096 waves; each wave streams a contiguous
    // run of ~32 tiles, 64 loads in flight per iteration.
    const int threads = 256;
    int blocks = 512;
    const int maxBlocks = (nTiles + 7) / 8;       // >= 1 tile per wave
    if (blocks > maxBlocks && maxBlocks > 0) blocks = maxBlocks;

    ripple_add_wmma_kernel<<<blocks, threads, 0, stream>>>(a, b, out, N, nTiles);
}
